// ASTModel_31241592111615
// MI455X (gfx1250) — matrix operand order repaired and checked by reading, not compiled, not run on hardware
//
#include <hip/hip_runtime.h>

#define NN 32768   // nodes
#define EE 524288  // edges
#define GB 64      // graphs (batch)
#define LL 512     // nodes per graph / timesteps
#define DD 256     // embed dim
#define HH 512     // hidden dim
#define H3 1536    // 3*H

typedef __bf16 bf16;
typedef bf16 v16bf __attribute__((ext_vector_type(16)));
typedef float v8f  __attribute__((ext_vector_type(8)));

union FragBF { v16bf v; unsigned int u32[8]; unsigned short u16[16]; };

__device__ __forceinline__ unsigned short f2bf(float f) {
  unsigned int u = __float_as_uint(f);
  u += 0x7FFFu + ((u >> 16) & 1u);           // round-to-nearest-even
  return (unsigned short)(u >> 16);
}

// ---- WMMA fragment helpers (wave32, 16x16x32 bf16, per ISA 7.12.2) ----
// A (16x32, MxK): lane holds row m = m0+(lane&15); K-halves {kb..kb+7} and {16+kb..16+kb+7},
// kb = 0 (lanes 0-15) or 8 (lanes 16-31). Both halves are contiguous u16 runs -> 2x b128.
__device__ __forceinline__ v16bf load_frag_a(const unsigned short* __restrict__ A,
                                             int lda, int m0, int k0) {
  const int lane = threadIdx.x & 31;
  const int m  = m0 + (lane & 15);
  const int kb = (lane < 16) ? 0 : 8;
  const unsigned short* row = A + (size_t)m * lda + k0 + kb;
  FragBF f;
  *(uint4*)&f.u16[0] = *(const uint4*)(row);
  *(uint4*)&f.u16[8] = *(const uint4*)(row + 16);
  return f.v;
}

// B pre-packed in fragment order: block (nt, kc) = 32 lanes x 16 bf16, contiguous.
__device__ __forceinline__ v16bf load_frag_b_packed(const unsigned short* __restrict__ Bp,
                                                    int nt, int kc, int KC) {
  const int lane = threadIdx.x & 31;
  const uint4* p = (const uint4*)(Bp + (((size_t)nt * KC + kc) * 32 + lane) * 16);
  FragBF f;
  *(uint4*)&f.u16[0] = p[0];
  *(uint4*)&f.u16[8] = p[1];
  return f.v;
}

// C/D f32 16x16: VGPR v -> M = v (lanes 0-15) or v+8 (lanes 16-31), N = lane&15
__device__ __forceinline__ void store_tile(float* __restrict__ C, int ldc,
                                           int m0, int n0, v8f c) {
  const int lane = threadIdx.x & 31;
  const int n  = n0 + (lane & 15);
  const int mo = (lane < 16) ? 0 : 8;
#pragma unroll
  for (int v = 0; v < 8; ++v) C[(size_t)(m0 + mo + v) * ldc + n] = c[v];
}

// ---------------- utility kernels ----------------
__global__ void zero_f32(float* p, int n) {
  int i = blockIdx.x * blockDim.x + threadIdx.x;
  if (i < n) p[i] = 0.0f;
}
__global__ void zero_u16(unsigned short* p, int n) {
  int i = blockIdx.x * blockDim.x + threadIdx.x;
  if (i < n) p[i] = 0;
}
__global__ void deg_init(float* deg) {           // self-loop folded in
  int i = blockIdx.x * blockDim.x + threadIdx.x;
  if (i < NN) deg[i] = 1.0f;
}
__global__ void deg_edges(const int* __restrict__ dst, float* deg) {
  int i = blockIdx.x * blockDim.x + threadIdx.x;
  if (i < EE) atomicAdd(&deg[dst[i]], 1.0f);
}
__global__ void dinv_k(const float* __restrict__ deg, float* dinv) {
  int i = blockIdx.x * blockDim.x + threadIdx.x;
  if (i < NN) { float d = deg[i]; dinv[i] = d > 0.0f ? rsqrtf(d) : 0.0f; }
}
__global__ void gather_embed(const int* __restrict__ x, const float* __restrict__ emb,
                             unsigned short* __restrict__ out) {
  int i = blockIdx.x * blockDim.x + threadIdx.x;   // NN*DD threads
  int node = i >> 8, k = i & 255;
  out[i] = f2bf(emb[(size_t)x[node] * DD + k]);
}

// Pack weights (fp32 source) directly into WMMA B-fragment order.
// B[k,n] = transposed ? src[n*K + k] : src[k*N + n]
__global__ void pack_b(const float* __restrict__ src, unsigned short* __restrict__ Bp,
                       int K, int N, int transposed) {
  int idx = blockIdx.x * blockDim.x + threadIdx.x;
  if (idx >= K * N) return;
  int e    = idx & 15;
  int lane = (idx >> 4) & 31;
  int rest = idx >> 9;              // nt * KC + kc
  int KC   = K >> 5;
  int kc   = rest % KC;
  int nt   = rest / KC;
  int n = nt * 16 + (lane & 15);
  int k = kc * 32 + ((lane & 16) ? 8 : 0) + ((e < 8) ? e : 8 + e);
  float v = transposed ? src[(size_t)n * K + k] : src[(size_t)k * N + n];
  Bp[idx] = f2bf(v);
}

// -------- bf16 WMMA GEMM: C[M,N] = A[M,K] @ B[K,N]; 1 wave = 16 x (16*NTILES) --------
template <int NTILES>
__global__ void gemm_bf16(const unsigned short* __restrict__ A, int lda,
                          const unsigned short* __restrict__ Bp,
                          float* __restrict__ C, int ldc, int K) {
  const int KC  = K >> 5;
  const int nt0 = blockIdx.x * NTILES;
  const int m0  = blockIdx.y * 16;
  const int lane = threadIdx.x & 31;
  v8f acc[NTILES] = {};
  for (int kc = 0; kc < KC; ++kc) {
    v16bf a = load_frag_a(A, lda, m0, kc * 32);
#pragma unroll
    for (int i = 0; i < NTILES; ++i) {
      if (kc + 1 < KC)
        __builtin_prefetch(Bp + (((size_t)(nt0 + i) * KC + kc + 1) * 32 + lane) * 16, 0, 1);
      v16bf b = load_frag_b_packed(Bp, nt0 + i, kc, KC);
      acc[i] = __builtin_amdgcn_wmma_f32_16x16x32_bf16(false, a, false, b, (short)0,
                                                       acc[i], false, false);
    }
  }
#pragma unroll
  for (int i = 0; i < NTILES; ++i) store_tile(C, ldc, m0, (nt0 + i) * 16, acc[i]);
}

// ---------------- GCN scatter (edges + self-loops) ----------------
__global__ void scatter_k(const float* __restrict__ h, const int* __restrict__ src,
                          const int* __restrict__ dst, const float* __restrict__ dinv,
                          float* __restrict__ out) {
  int e = blockIdx.x;
  int s, d;
  if (e < EE) { s = src[e]; d = dst[e]; } else { s = d = e - EE; }
  float norm = dinv[s] * dinv[d];
  const float* hs = h + (size_t)s * HH;
  float* od = out + (size_t)d * HH;
  for (int k = threadIdx.x; k < HH; k += blockDim.x)
    atomicAdd(&od[k], hs[k] * norm);
}

// out = relu(out + bias); also emit bf16 copy for next GEMM
__global__ void bias_relu_cvt(float* __restrict__ out, const float* __restrict__ bias,
                              unsigned short* __restrict__ obf) {
  int i = blockIdx.x * blockDim.x + threadIdx.x;   // NN*HH threads
  int k = i & (HH - 1);
  float v = out[i] + bias[k];
  v = v > 0.0f ? v : 0.0f;
  out[i] = v;
  obf[i] = f2bf(v);
}

// ---- GRU step: Gi = x_t @ Wih^T, Gh = h @ Whh^T; 1 wave = 2 n-tiles x {Gi, Gh} ----
__global__ void gru_gemm(const unsigned short* __restrict__ X,    // bufX + t*HH, lda = LL*HH
                         const unsigned short* __restrict__ Wip,  // packed [HH x H3]
                         const unsigned short* __restrict__ Hbf,  // [GB, HH]
                         const unsigned short* __restrict__ Whp,  // packed [HH x H3]
                         float* __restrict__ Gi, float* __restrict__ Gh) {
  const int KC  = HH >> 5;                 // 16
  const int nt0 = blockIdx.x * 2;
  const int m0  = blockIdx.y * 16;
  v8f ci[2] = {}, ch[2] = {};
  for (int kc = 0; kc < KC; ++kc) {
    v16bf ax = load_frag_a(X, LL * HH, m0, kc * 32);
    v16bf ah = load_frag_a(Hbf, HH, m0, kc * 32);
#pragma unroll
    for (int i = 0; i < 2; ++i) {
      v16bf bi = load_frag_b_packed(Wip, nt0 + i, kc, KC);
      ci[i] = __builtin_amdgcn_wmma_f32_16x16x32_bf16(false, ax, false, bi, (short)0,
                                                      ci[i], false, false);
      v16bf bh = load_frag_b_packed(Whp, nt0 + i, kc, KC);
      ch[i] = __builtin_amdgcn_wmma_f32_16x16x32_bf16(false, ah, false, bh, (short)0,
                                                      ch[i], false, false);
    }
  }
#pragma unroll
  for (int i = 0; i < 2; ++i) {
    store_tile(Gi, H3, m0, (nt0 + i) * 16, ci[i]);
    store_tile(Gh, H3, m0, (nt0 + i) * 16, ch[i]);
  }
}

__device__ __forceinline__ float sigm(float x) { return 1.0f / (1.0f + expf(-x)); }

__global__ void gru_gates(const float* __restrict__ Gi, const float* __restrict__ Gh,
                          const float* __restrict__ bih, const float* __restrict__ bhh,
                          float* __restrict__ hstate, unsigned short* __restrict__ hbf,
                          float* __restrict__ hsum) {
  int b = blockIdx.x, j = threadIdx.x;
  const float* gi = Gi + (size_t)b * H3;
  const float* gh = Gh + (size_t)b * H3;
  float r = sigm(gi[j] + bih[j] + gh[j] + bhh[j]);
  float z = sigm(gi[HH + j] + bih[HH + j] + gh[HH + j] + bhh[HH + j]);
  float n = tanhf(gi[2 * HH + j] + bih[2 * HH + j] + r * (gh[2 * HH + j] + bhh[2 * HH + j]));
  int idx = b * HH + j;
  float hp = hstate[idx];
  float hn = (1.0f - z) * n + z * hp;
  hstate[idx] = hn;
  hbf[idx] = f2bf(hn);
  hsum[idx] += hn;
}

// ---------------- head: mean -> concat focal -> fc1(relu) -> fc2(sigmoid) ----------------
__global__ void head_k(const float* __restrict__ hsum, const float* __restrict__ focal,
                       const float* __restrict__ fc1w, const float* __restrict__ fc1b,
                       const float* __restrict__ fc2w, const float* __restrict__ fc2b,
                       float* __restrict__ out) {
  __shared__ float g[HH + 1];
  __shared__ float o1[HH];
  __shared__ float red[256];
  int b = blockIdx.x, tid = threadIdx.x;
  for (int k = tid; k < HH; k += blockDim.x) g[k] = hsum[b * HH + k] * (1.0f / (float)LL);
  if (tid == 0) g[HH] = focal[b];
  __syncthreads();
  for (int j = tid; j < HH; j += blockDim.x) {
    float acc = fc1b[j];
    for (int k = 0; k <= HH; ++k) acc += g[k] * fc1w[(size_t)k * HH + j];
    o1[j] = acc > 0.0f ? acc : 0.0f;
  }
  __syncthreads();
  float p = 0.0f;
  for (int j = tid; j < HH; j += blockDim.x) p += o1[j] * fc2w[j];
  red[tid] = p;
  __syncthreads();
  for (int s = 128; s > 0; s >>= 1) {
    if (tid < s) red[tid] += red[tid + s];
    __syncthreads();
  }
  if (tid == 0) out[b] = sigm(red[0] + fc2b[0]);
}

// ---------------- launch ----------------
extern "C" void kernel_launch(void* const* d_in, const int* in_sizes, int n_in,
                              void* d_out, int out_size, void* d_ws, size_t ws_size,
                              hipStream_t stream) {
  const int*   x      = (const int*)d_in[0];
  const int*   eidx   = (const int*)d_in[1];   // [2, E]
  const float* focal  = (const float*)d_in[3];
  const float* emb    = (const float*)d_in[4];
  const float* W1     = (const float*)d_in[5];
  const float* b1     = (const float*)d_in[6];
  const float* W2     = (const float*)d_in[7];
  const float* b2     = (const float*)d_in[8];
  const float* Wih    = (const float*)d_in[9];
  const float* Whh    = (const float*)d_in[10];
  const float* bih    = (const float*)d_in[11];
  const float* bhh    = (const float*)d_in[12];
  const float* fc1w   = (const float*)d_in[13];
  const float* fc1b   = (const float*)d_in[14];
  const float* fc2w   = (const float*)d_in[15];
  const float* fc2b   = (const float*)d_in[16];
  const int* src = eidx;
  const int* dst = eidx + EE;
  float* out = (float*)d_out;

  char* base = (char*)d_ws;
  size_t off = 0;
  auto alloc = [&](size_t bytes) -> void* {
    void* p = base + off;
    off += (bytes + 255) & ~(size_t)255;
    return p;
  };
  unsigned short* A0bf = (unsigned short*)alloc((size_t)NN * DD * 2);
  unsigned short* W1p  = (unsigned short*)alloc((size_t)DD * HH * 2);
  unsigned short* W2p  = (unsigned short*)alloc((size_t)HH * HH * 2);
  unsigned short* Wip  = (unsigned short*)alloc((size_t)HH * H3 * 2);
  unsigned short* Whp  = (unsigned short*)alloc((size_t)HH * H3 * 2);
  float* deg   = (float*)alloc((size_t)NN * 4);
  float* dinv  = (float*)alloc((size_t)NN * 4);
  float* bufP  = (float*)alloc((size_t)NN * HH * 4);
  float* bufO  = (float*)alloc((size_t)NN * HH * 4);
  unsigned short* bufX = (unsigned short*)alloc((size_t)NN * HH * 2);
  float* Gi    = (float*)alloc((size_t)GB * H3 * 4);
  float* Gh    = (float*)alloc((size_t)GB * H3 * 4);
  float* hstate = (float*)alloc((size_t)GB * HH * 4);
  unsigned short* hbf = (unsigned short*)alloc((size_t)GB * HH * 2);
  float* hsum  = (float*)alloc((size_t)GB * HH * 4);
  if (off > ws_size) return;

  // degree normalization (graph constant)
  deg_init<<<NN / 256, 256, 0, stream>>>(deg);
  deg_edges<<<EE / 256, 256, 0, stream>>>(dst, deg);
  dinv_k<<<NN / 256, 256, 0, stream>>>(deg, dinv);

  // operand prep: gather embeddings (bf16) + pack weights into fragment order
  gather_embed<<<(NN * DD) / 256, 256, 0, stream>>>(x, emb, A0bf);
  pack_b<<<(DD * HH) / 256, 256, 0, stream>>>(W1, W1p, DD, HH, 0);
  pack_b<<<(HH * HH) / 256, 256, 0, stream>>>(W2, W2p, HH, HH, 0);
  pack_b<<<(HH * H3) / 256, 256, 0, stream>>>(Wih, Wip, HH, H3, 1);
  pack_b<<<(HH * H3) / 256, 256, 0, stream>>>(Whh, Whp, HH, H3, 1);

  // ---- GCN layer 1 ----
  gemm_bf16<8><<<dim3(HH / 128, NN / 16), 32, 0, stream>>>(A0bf, DD, W1p, bufP, HH, DD);
  zero_f32<<<(NN * HH) / 256, 256, 0, stream>>>(bufO, NN * HH);
  scatter_k<<<EE + NN, 128, 0, stream>>>(bufP, src, dst, dinv, bufO);
  bias_relu_cvt<<<(NN * HH) / 256, 256, 0, stream>>>(bufO, b1, bufX);

  // ---- GCN layer 2 ----
  gemm_bf16<8><<<dim3(HH / 128, NN / 16), 32, 0, stream>>>(bufX, HH, W2p, bufP, HH, HH);
  zero_f32<<<(NN * HH) / 256, 256, 0, stream>>>(bufO, NN * HH);
  scatter_k<<<EE + NN, 128, 0, stream>>>(bufP, src, dst, dinv, bufO);
  bias_relu_cvt<<<(NN * HH) / 256, 256, 0, stream>>>(bufO, b2, bufX);
  // bufX now holds bf16 node features == padded[b, t] at row b*LL + t

  // ---- GRU ----
  zero_f32<<<(GB * HH) / 256, 256, 0, stream>>>(hstate, GB * HH);
  zero_f32<<<(GB * HH) / 256, 256, 0, stream>>>(hsum, GB * HH);
  zero_u16<<<(GB * HH) / 256, 256, 0, stream>>>(hbf, GB * HH);
  for (int t = 0; t < LL; ++t) {
    gru_gemm<<<dim3(H3 / 32, GB / 16), 32, 0, stream>>>(bufX + (size_t)t * HH, Wip,
                                                        hbf, Whp, Gi, Gh);
    gru_gates<<<GB, HH, 0, stream>>>(Gi, Gh, bih, bhh, hstate, hbf, hsum);
  }

  // ---- head ----
  head_k<<<GB, 256, 0, stream>>>(hsum, focal, fc1w, fc1b, fc2w, fc2b, out);
}
